// SlotAttention_74088185856596
// MI455X (gfx1250) — matrix operand order repaired and checked by reading, not compiled, not run on hardware
//
#include <hip/hip_runtime.h>

typedef __attribute__((ext_vector_type(16))) _Float16 v16h;
typedef __attribute__((ext_vector_type(2)))  _Float16 h2;
typedef __attribute__((ext_vector_type(8)))  float    v8f;

#define DEV __device__ __forceinline__

// ---- WMMA fragment helpers (CDNA5 16x16x32 f16, wave32) --------------------
// A (16xK tile, row-major, leading dim ld): lane = M (0..15), lanes 16-31 take
// K offset +8; VGPR v holds K pair at ((v>>2)*16 + (v&3)*2).
DEV v16h ld_fragA(const _Float16* p, int ld) {
  const int lane = threadIdx.x & 31;
  const _Float16* rp = p + (size_t)(lane & 15) * ld + ((lane >> 4) << 3);
  const h2* rp2 = (const h2*)rp;
  v16h f; h2* fp = (h2*)&f;
#pragma unroll
  for (int v = 0; v < 8; ++v) fp[v] = rp2[((v >> 2) << 3) + (v & 3)];
  return f;
}
// B stored transposed (Bt[n][k], row-major, leading dim ld): lane = N (0..15),
// lanes 16-31 take K offset +8; VGPR v holds K pair at ((v>>2)*16 + (v&3)*2).
DEV v16h ld_fragB(const _Float16* p, int ld) {
  const int lane = threadIdx.x & 31;
  const _Float16* rp = p + (size_t)(lane & 15) * ld + ((lane >> 4) << 3);
  const h2* rp2 = (const h2*)rp;
  v16h f; h2* fp = (h2*)&f;
#pragma unroll
  for (int v = 0; v < 8; ++v) fp[v] = rp2[((v >> 2) << 3) + (v & 3)];
  return f;
}
DEV v8f wmma_f16(const v16h& a, const v16h& b, const v8f& c) {
  return __builtin_amdgcn_wmma_f32_16x16x32_f16(false, a, false, b, (short)0, c,
                                                false, false);
}
DEV void zero_acc(v8f* acc) {
#pragma unroll
  for (int t = 0; t < 4; ++t)
#pragma unroll
    for (int r = 0; r < 8; ++r) acc[t][r] = 0.f;
}

// ---- small utility kernels -------------------------------------------------
__global__ void zero_h_k(_Float16* p, int n) {
  for (int i = blockIdx.x * blockDim.x + threadIdx.x; i < n;
       i += gridDim.x * blockDim.x)
    p[i] = (_Float16)0.f;
}
__global__ void copy_f_k(float* dst, const float* src, int n) {
  for (int i = blockIdx.x * blockDim.x + threadIdx.x; i < n;
       i += gridDim.x * blockDim.x)
    dst[i] = src[i];
}

// LN over F=768, one row per block (256 threads, 3 elems each) -> f16
__global__ void ln_in_k(const float* __restrict__ inp, const float* __restrict__ w,
                        const float* __restrict__ b, _Float16* __restrict__ xln) {
  __shared__ float s1[256], s2[256];
  const size_t row = blockIdx.x;
  const int tid = threadIdx.x;
  const float* rp = inp + row * 768;
  float x0 = rp[tid], x1 = rp[tid + 256], x2 = rp[tid + 512];
  s1[tid] = x0 + x1 + x2;
  s2[tid] = x0 * x0 + x1 * x1 + x2 * x2;
  __syncthreads();
  for (int o = 128; o; o >>= 1) {
    if (tid < o) { s1[tid] += s1[tid + o]; s2[tid] += s2[tid + o]; }
    __syncthreads();
  }
  float m = s1[0] * (1.f / 768.f);
  float var = s2[0] * (1.f / 768.f) - m * m;
  float rstd = rsqrtf(var + 1e-5f);
  _Float16* op = xln + row * 768;
  op[tid]       = (_Float16)((x0 - m) * rstd * w[tid]       + b[tid]);
  op[tid + 256] = (_Float16)((x1 - m) * rstd * w[tid + 256] + b[tid + 256]);
  op[tid + 512] = (_Float16)((x2 - m) * rstd * w[tid + 512] + b[tid + 512]);
}

// LN over D=256 for slots, one row per block; also emits f16 copy of raw slots
__global__ void ln_slots_k(const float* __restrict__ slots, const float* __restrict__ w,
                           const float* __restrict__ b, _Float16* __restrict__ sn16,
                           _Float16* __restrict__ s16) {
  __shared__ float s1[256], s2[256];
  const int row = blockIdx.x, tid = threadIdx.x;
  float x = slots[row * 256 + tid];
  s1[tid] = x; s2[tid] = x * x;
  __syncthreads();
  for (int o = 128; o; o >>= 1) {
    if (tid < o) { s1[tid] += s1[tid + o]; s2[tid] += s2[tid + o]; }
    __syncthreads();
  }
  float m = s1[0] * (1.f / 256.f);
  float var = s2[0] * (1.f / 256.f) - m * m;
  float rstd = rsqrtf(var + 1e-5f);
  sn16[row * 256 + tid] = (_Float16)((x - m) * rstd * w[tid] + b[tid]);
  s16[row * 256 + tid] = (_Float16)x;
}

// Convert weights to f16: wkv16 = [Wk;Wv] (512x768), wq16 (256x256),
// wih16 (768x256), whh16 (768x256). All keep X@W.T row-major layout (= Bt).
__global__ void cvt_w_k(const float* __restrict__ Wq, const float* __restrict__ Wk,
                        const float* __restrict__ Wv, const float* __restrict__ Wih,
                        const float* __restrict__ Whh, _Float16* wkv16,
                        _Float16* wq16, _Float16* wih16, _Float16* whh16) {
  int i = blockIdx.x * 256 + threadIdx.x;
  if (i < 393216) {
    int j = i / 768, f = i - j * 768;
    float v = (j < 256) ? Wk[j * 768 + f] : Wv[(size_t)(j - 256) * 768 + f];
    wkv16[i] = (_Float16)v;
  } else if (i < 458752) {
    int t = i - 393216; wq16[t] = (_Float16)Wq[t];
  } else if (i < 655360) {
    int t = i - 458752; wih16[t] = (_Float16)Wih[t];
  } else if (i < 851968) {
    int t = i - 655360; whh16[t] = (_Float16)Whh[t];
  }
}

// ---- GEMM 1: [65536,768] @ wkv^T -> k (f16 [B][N][256]) and vT (f16 [B][256][N])
__global__ __launch_bounds__(256) void gemm_kv_k(
    const _Float16* __restrict__ X, const _Float16* __restrict__ W,
    _Float16* __restrict__ Kout, _Float16* __restrict__ VT) {
  int wid = (int)((blockIdx.x * blockDim.x + threadIdx.x) >> 5);
  int wm = wid >> 3, wn = wid & 7;
  size_t arow = (size_t)wm * 16;
  v8f acc[4]; zero_acc(acc);
  const _Float16* Abase = X + arow * 768;
  for (int kt = 0; kt < 768; kt += 32) {
    v16h a = ld_fragA(Abase + kt, 768);
#pragma unroll
    for (int t = 0; t < 4; ++t) {
      v16h bf = ld_fragB(W + (size_t)(wn * 64 + t * 16) * 768 + kt, 768);
      acc[t] = wmma_f16(a, bf, acc[t]);
    }
  }
  int lane = threadIdx.x & 31;
  int nn = lane & 15, mo = (lane >> 4) << 3;
#pragma unroll
  for (int t = 0; t < 4; ++t) {
    int j = wn * 64 + t * 16 + nn;
#pragma unroll
    for (int r = 0; r < 8; ++r) {
      size_t row = arow + mo + r;
      float val = acc[t][r];
      if (j < 256) {
        Kout[row * 256 + j] = (_Float16)val;
      } else {
        size_t bb = row >> 10, ncol = row & 1023;
        VT[(bb * 256 + (size_t)(j - 256)) * 1024 + ncol] = (_Float16)val;
      }
    }
  }
}

// ---- GEMM q: [512,256] @ Wq^T * SCALE -> q16 [B][H][16(pad)][64]
__global__ __launch_bounds__(256) void gemm_q_k(
    const _Float16* __restrict__ SN, const _Float16* __restrict__ WQ,
    _Float16* __restrict__ Q16) {
  int wid = (int)((blockIdx.x * blockDim.x + threadIdx.x) >> 5);
  int wm = wid >> 2, wn = wid & 3;  // wn == head
  v8f acc[4]; zero_acc(acc);
  for (int kt = 0; kt < 256; kt += 32) {
    v16h a = ld_fragA(SN + (size_t)wm * 16 * 256 + kt, 256);
#pragma unroll
    for (int t = 0; t < 4; ++t) {
      v16h bf = ld_fragB(WQ + (size_t)(wn * 64 + t * 16) * 256 + kt, 256);
      acc[t] = wmma_f16(a, bf, acc[t]);
    }
  }
  int lane = threadIdx.x & 31;
  int nn = lane & 15, mo = (lane >> 4) << 3;
#pragma unroll
  for (int t = 0; t < 4; ++t)
#pragma unroll
    for (int r = 0; r < 8; ++r) {
      int row = wm * 16 + mo + r;  // 0..511
      int b = row >> 3, s = row & 7;
      int d = t * 16 + nn;
      Q16[(size_t)((b * 4 + wn) * 16 + s) * 64 + d] = (_Float16)(acc[t][r] * 0.125f);
    }
}

// ---- GEMM dots: per (b,h): q[16pad,64] @ k_h^T -> dots f32 [B][S][H][N]
__global__ __launch_bounds__(256) void gemm_dots_k(
    const _Float16* __restrict__ Q16, const _Float16* __restrict__ K16,
    float* __restrict__ dots) {
  int wid = (int)((blockIdx.x * blockDim.x + threadIdx.x) >> 5);
  int nt = wid & 15, h = (wid >> 4) & 3, b = wid >> 6;
  const _Float16* A = Q16 + (size_t)((b * 4 + h) * 16) * 64;
  const _Float16* Bb = K16 + (size_t)b * 1024 * 256 + h * 64;
  int nbase = nt * 64;
  v8f acc[4]; zero_acc(acc);
  for (int kt = 0; kt < 64; kt += 32) {
    v16h a = ld_fragA(A + kt, 64);
#pragma unroll
    for (int t = 0; t < 4; ++t) {
      v16h bf = ld_fragB(Bb + (size_t)(nbase + t * 16) * 256 + kt, 256);
      acc[t] = wmma_f16(a, bf, acc[t]);
    }
  }
  int lane = threadIdx.x & 31;
  int nn = lane & 15, mo = (lane >> 4) << 3;
#pragma unroll
  for (int t = 0; t < 4; ++t)
#pragma unroll
    for (int r = 0; r < 8; ++r) {
      int s = mo + r;
      if (s < 8)
        dots[(size_t)((b * 8 + s) * 4 + h) * 1024 + nbase + t * 16 + nn] = acc[t][r];
    }
}

// softmax over the 32 = (s,h) axis for each (b, n), in place
__global__ void softmax32_k(float* __restrict__ dots) {
  int idx = blockIdx.x * 256 + threadIdx.x;  // 65536 = B*N
  int b = idx >> 10, n = idx & 1023;
  float* col = dots + (size_t)b * 32768 + n;
  float e[32], mx = -3.4e38f;
#pragma unroll
  for (int i = 0; i < 32; ++i) { e[i] = col[(size_t)i * 1024]; mx = fmaxf(mx, e[i]); }
  float s = 0.f;
#pragma unroll
  for (int i = 0; i < 32; ++i) { e[i] = expf(e[i] - mx); s += e[i]; }
  float inv = 1.f / s;
#pragma unroll
  for (int i = 0; i < 32; ++i) col[(size_t)i * 1024] = e[i] * inv;
}

// attn_w = attn / (sum_n attn + eps) -> f16 [B][H][16(pad)][N]
__global__ void rownorm_k(const float* __restrict__ dots, _Float16* __restrict__ aw16) {
  __shared__ float red[256];
  int row = blockIdx.x, tid = threadIdx.x;  // row = b*32 + s*4 + h
  const float* r = dots + (size_t)row * 1024;
  red[tid] = r[tid] + r[tid + 256] + r[tid + 512] + r[tid + 768];
  __syncthreads();
  for (int o = 128; o; o >>= 1) {
    if (tid < o) red[tid] += red[tid + o];
    __syncthreads();
  }
  float inv = 1.f / (red[0] + 1e-8f);
  int b = row >> 5, s = (row >> 2) & 7, h = row & 3;
  _Float16* w = aw16 + (size_t)((b * 4 + h) * 16 + s) * 1024;
  for (int i = tid; i < 1024; i += 256) w[i] = (_Float16)(r[i] * inv);
}

// attn_vis = mean over heads (last iteration only)
__global__ void attn_vis_k(const float* __restrict__ dots, float* __restrict__ out) {
  int idx = blockIdx.x * 256 + threadIdx.x;  // 524288 = B*S*N
  int b = idx >> 13, rem = idx & 8191, s = rem >> 10, n = rem & 1023;
  const float* dp = dots + (size_t)((b * 8 + s) * 4) * 1024 + n;
  out[idx] = 0.25f * (dp[0] + dp[1024] + dp[2048] + dp[3072]);
}

// ---- GEMM updates: per (b,h): attn_w[16pad,1024] @ vT_h^T -> u16 [512][256]
__global__ __launch_bounds__(256) void gemm_upd_k(
    const _Float16* __restrict__ AW, const _Float16* __restrict__ VT,
    _Float16* __restrict__ U16) {
  int wid = (int)((blockIdx.x * blockDim.x + threadIdx.x) >> 5);  // 0..255
  int b = wid >> 2, h = wid & 3;
  const _Float16* A = AW + (size_t)((b * 4 + h) * 16) * 1024;
  const _Float16* Bb = VT + (size_t)(b * 256 + h * 64) * 1024;
  v8f acc[4]; zero_acc(acc);
  for (int kt = 0; kt < 1024; kt += 32) {
    v16h a = ld_fragA(A + kt, 1024);
#pragma unroll
    for (int t = 0; t < 4; ++t) {
      v16h bf = ld_fragB(Bb + (size_t)(t * 16) * 1024 + kt, 1024);
      acc[t] = wmma_f16(a, bf, acc[t]);
    }
  }
  int lane = threadIdx.x & 31;
  int nn = lane & 15, mo = (lane >> 4) << 3;
#pragma unroll
  for (int t = 0; t < 4; ++t)
#pragma unroll
    for (int r = 0; r < 8; ++r) {
      int s = mo + r;
      if (s < 8)
        U16[(size_t)(b * 8 + s) * 256 + h * 64 + t * 16 + nn] = (_Float16)acc[t][r];
    }
}

// ---- GRU gate GEMMs: gi = u16 @ Wih^T, gh = s16 @ Whh^T (both [512,768])
__global__ __launch_bounds__(256) void gemm_gru_k(
    const _Float16* __restrict__ U16, const _Float16* __restrict__ S16,
    const _Float16* __restrict__ WIH, const _Float16* __restrict__ WHH,
    float* __restrict__ gi, float* __restrict__ gh) {
  int wid = (int)((blockIdx.x * blockDim.x + threadIdx.x) >> 5);  // 0..767
  int g = wid / 384, rem = wid - g * 384;
  int wm = rem / 12, wn = rem - wm * 12;
  const _Float16* A = g ? S16 : U16;
  const _Float16* W = g ? WHH : WIH;
  float* Out = g ? gh : gi;
  v8f acc[4]; zero_acc(acc);
  for (int kt = 0; kt < 256; kt += 32) {
    v16h a = ld_fragA(A + (size_t)wm * 16 * 256 + kt, 256);
#pragma unroll
    for (int t = 0; t < 4; ++t) {
      v16h bf = ld_fragB(W + (size_t)(wn * 64 + t * 16) * 256 + kt, 256);
      acc[t] = wmma_f16(a, bf, acc[t]);
    }
  }
  int lane = threadIdx.x & 31;
  int nn = lane & 15, mo = (lane >> 4) << 3;
#pragma unroll
  for (int t = 0; t < 4; ++t)
#pragma unroll
    for (int r = 0; r < 8; ++r)
      Out[(size_t)(wm * 16 + mo + r) * 768 + wn * 64 + t * 16 + nn] = acc[t][r];
}

// GRU pointwise update (in-place on slots f32)
__global__ void gru_cell_k(const float* __restrict__ gi, const float* __restrict__ gh,
                           const float* __restrict__ bih, const float* __restrict__ bhh,
                           float* __restrict__ slots) {
  int idx = blockIdx.x * 256 + threadIdx.x;  // 0..131071
  int row = idx >> 8, d = idx & 255;
  const float* gir = gi + (size_t)row * 768;
  const float* ghr = gh + (size_t)row * 768;
  float ir = gir[d] + bih[d],            hr = ghr[d] + bhh[d];
  float iz = gir[256 + d] + bih[256 + d], hz = ghr[256 + d] + bhh[256 + d];
  float in_ = gir[512 + d] + bih[512 + d], hn = ghr[512 + d] + bhh[512 + d];
  float r = 1.f / (1.f + expf(-(ir + hr)));
  float z = 1.f / (1.f + expf(-(iz + hz)));
  float nn = tanhf(in_ + r * hn);
  float h = slots[idx];
  slots[idx] = (1.f - z) * nn + z * h;
}

// ---------------------------------------------------------------------------
extern "C" void kernel_launch(void* const* d_in, const int* in_sizes, int n_in,
                              void* d_out, int out_size, void* d_ws, size_t ws_size,
                              hipStream_t stream) {
  const float* inputs  = (const float*)d_in[0];
  const float* slots_in= (const float*)d_in[1];
  const float* Wq      = (const float*)d_in[2];
  const float* Wk      = (const float*)d_in[3];
  const float* Wv      = (const float*)d_in[4];
  const float* ln_in_w = (const float*)d_in[5];
  const float* ln_in_b = (const float*)d_in[6];
  const float* ln_s_w  = (const float*)d_in[7];
  const float* ln_s_b  = (const float*)d_in[8];
  const float* Wih     = (const float*)d_in[9];
  const float* Whh     = (const float*)d_in[10];
  const float* bih     = (const float*)d_in[11];
  const float* bhh     = (const float*)d_in[12];
  float* out = (float*)d_out;

  char* p = (char*)d_ws;
  auto take = [&](size_t bytes) {
    char* r = p;
    p += (bytes + 255) & ~(size_t)255;
    return r;
  };
  _Float16* xln16 = (_Float16*)take((size_t)65536 * 768 * 2);   // LN(inputs) f16
  _Float16* wkv16 = (_Float16*)take((size_t)512 * 768 * 2);     // [Wk;Wv] f16
  _Float16* wq16  = (_Float16*)take((size_t)65536 * 2);
  _Float16* wih16 = (_Float16*)take((size_t)196608 * 2);
  _Float16* whh16 = (_Float16*)take((size_t)196608 * 2);
  _Float16* k16   = (_Float16*)take((size_t)65536 * 256 * 2);   // k [B][N][256]
  _Float16* vT16  = (_Float16*)take((size_t)64 * 256 * 1024 * 2); // v^T [B][256][N]
  _Float16* q16   = (_Float16*)take((size_t)65536 * 2);         // [B][H][16][64]
  _Float16* sn16  = (_Float16*)take((size_t)131072 * 2);
  _Float16* s16   = (_Float16*)take((size_t)131072 * 2);
  _Float16* aw16  = (_Float16*)take((size_t)64 * 4 * 16 * 1024 * 2); // attn_w
  _Float16* u16   = (_Float16*)take((size_t)131072 * 2);        // updates f16
  float* dots  = (float*)take((size_t)64 * 32 * 1024 * 4);
  float* gi    = (float*)take((size_t)512 * 768 * 4);
  float* gh    = (float*)take((size_t)512 * 768 * 4);
  float* slots = (float*)take((size_t)131072 * 4);

  // init: zero padded tiles once (pad rows are never written later), copy slots
  zero_h_k<<<256, 256, 0, stream>>>(q16, 65536);
  zero_h_k<<<4096, 256, 0, stream>>>(aw16, 4194304);
  copy_f_k<<<512, 256, 0, stream>>>(slots, slots_in, 131072);

  ln_in_k<<<65536, 256, 0, stream>>>(inputs, ln_in_w, ln_in_b, xln16);
  cvt_w_k<<<3328, 256, 0, stream>>>(Wq, Wk, Wv, Wih, Whh, wkv16, wq16, wih16, whh16);
  gemm_kv_k<<<4096, 256, 0, stream>>>(xln16, wkv16, k16, vT16);

  for (int it = 0; it < 3; ++it) {
    ln_slots_k<<<512, 256, 0, stream>>>(slots, ln_s_w, ln_s_b, sn16, s16);
    gemm_q_k<<<16, 256, 0, stream>>>(sn16, wq16, q16);
    gemm_dots_k<<<512, 256, 0, stream>>>(q16, k16, dots);
    softmax32_k<<<256, 256, 0, stream>>>(dots);
    rownorm_k<<<2048, 256, 0, stream>>>(dots, aw16);
    if (it == 2) attn_vis_k<<<2048, 256, 0, stream>>>(dots, out + 131072);
    gemm_upd_k<<<32, 256, 0, stream>>>(aw16, vT16, u16);
    gemm_gru_k<<<96, 256, 0, stream>>>(u16, s16, wih16, whh16, gi, gh);
    gru_cell_k<<<512, 256, 0, stream>>>(gi, gh, bih, bhh, slots);
  }
  copy_f_k<<<512, 256, 0, stream>>>(out, slots, 131072);

  (void)in_sizes; (void)n_in; (void)out_size; (void)ws_size;
}
